// HistoryAwarePredictionHead_47863115547171
// MI455X (gfx1250) — matrix operand order repaired and checked by reading, not compiled, not run on hardware
//
#include <hip/hip_runtime.h>
#include <hip/hip_bf16.h>

typedef __bf16 bf16_t;
typedef __attribute__((ext_vector_type(8)))  bf16_t v8bf;
typedef __attribute__((ext_vector_type(16))) bf16_t v16bf;
typedef __attribute__((ext_vector_type(8)))  float  v8f;

#define B_      2048
#define L_      200
#define DMODEL  512
#define HID     256
#define VOC     40000
#define VPAD    40448   // 79 * 512, multiple of block N-tile
#define LN_EPS  1e-5f

// ---------------------------------------------------------------------------
// Kernel 1: fused  h = GELU(LayerNorm(hidden @ W1 + b1))  -> bf16
// One block per batch row; 256 threads = 256 hidden outputs per row.
// ---------------------------------------------------------------------------
__global__ void __launch_bounds__(256)
mlp_ln_gelu_kernel(const float* __restrict__ hidden,
                   const float* __restrict__ W1,
                   const float* __restrict__ b1,
                   const float* __restrict__ gamma,
                   const float* __restrict__ beta,
                   bf16_t* __restrict__ hact)
{
    __shared__ float xrow[DMODEL];
    __shared__ float red[HID];

    const int row = blockIdx.x;
    const int j   = threadIdx.x;               // 0..255

    const float* hr = hidden + (size_t)row * DMODEL;
    xrow[j]       = hr[j];
    xrow[j + 256] = hr[j + 256];
    __syncthreads();

    // dot(hidden[row,:], W1[:,j]) ; W1 is (512,256) row-major -> coalesced over j
    float acc = b1[j];
#pragma unroll 4
    for (int k = 0; k < DMODEL; ++k)
        acc = fmaf(xrow[k], W1[(size_t)k * HID + j], acc);

    // LayerNorm over the 256 outputs of this row
    red[j] = acc;
    __syncthreads();
    for (int s = 128; s > 0; s >>= 1) {
        if (j < s) red[j] += red[j + s];
        __syncthreads();
    }
    const float mu = red[0] * (1.0f / HID);
    __syncthreads();

    const float d = acc - mu;
    red[j] = d * d;
    __syncthreads();
    for (int s = 128; s > 0; s >>= 1) {
        if (j < s) red[j] += red[j + s];
        __syncthreads();
    }
    const float var = red[0] * (1.0f / HID);

    float hn = d * rsqrtf(var + LN_EPS) * gamma[j] + beta[j];
    // exact GELU: x * 0.5 * (1 + erf(x / sqrt(2)))
    float g = 0.5f * hn * (1.0f + erff(hn * 0.70710678118654752f));

    hact[(size_t)row * HID + j] = (bf16_t)g;
}

// ---------------------------------------------------------------------------
// Kernel 2: transpose-convert W2 (256 x 40000 f32) -> W2t (VPAD x 256 bf16)
// Classic 32x32 LDS tile transpose; padded rows replicate column V-1 (never
// stored by the GEMM epilogue).
// ---------------------------------------------------------------------------
__global__ void __launch_bounds__(256)
w2_transpose_bf16_kernel(const float* __restrict__ W2,
                         bf16_t* __restrict__ w2t)
{
    __shared__ float tile[32][33];

    const int tx = threadIdx.x;                // 0..31
    const int ty = threadIdx.y;                // 0..7
    const int n0 = blockIdx.x * 32;
    const int k0 = blockIdx.y * 32;

    int n = n0 + tx;
    if (n >= VOC) n = VOC - 1;
#pragma unroll
    for (int iy = 0; iy < 4; ++iy) {
        const int k = k0 + ty + iy * 8;        // < 256 always
        tile[ty + iy * 8][tx] = W2[(size_t)k * VOC + n];
    }
    __syncthreads();

    const int k2 = k0 + tx;
#pragma unroll
    for (int iy = 0; iy < 4; ++iy) {
        const int n2 = n0 + ty + iy * 8;       // < VPAD by grid construction
        w2t[(size_t)n2 * HID + k2] = (bf16_t)tile[tx][ty + iy * 8];
    }
}

// ---------------------------------------------------------------------------
// Kernel 3: WMMA GEMM  out = (hact_bf16 @ W2t_bf16^T) * ls + b2 * ls
// Block: 256 threads = 8 wave32. Block tile = 64(M) x 512(N).
// Each wave computes a 64x64 sub-tile: 4 M-subtiles x 4 N-subtiles =
// 16 f32 accumulators. Per K-step (K=32): 8 fragment loads (16x b128)
// feed 16 WMMAs -> 1 vector load per v_wmma_f32_16x16x32_bf16.
// Fragment layouts per CDNA5 ISA 7.12.2 (16-bit A 16x32, B 32x16, f32 C/D).
// ---------------------------------------------------------------------------
__global__ void __launch_bounds__(256)
wmma_logits_kernel(const bf16_t* __restrict__ hact,   // (2048 x 256) row-major
                   const bf16_t* __restrict__ w2t,    // (VPAD x 256) n-major
                   const float*  __restrict__ b2,
                   const float*  __restrict__ ls_ptr,
                   float* __restrict__ out)
{
    const int m0   = blockIdx.y * 64;
    const int wave = threadIdx.x >> 5;          // 0..7
    const int lane = threadIdx.x & 31;
    const int half = lane >> 4;                 // 0 or 1
    const int l15  = lane & 15;
    const int n0   = blockIdx.x * 512 + wave * 64;

    const float ls = *ls_ptr;

    v8f acc[4][4] = {};                         // [m-subtile][n-subtile]

    const bf16_t* aBase = hact + (size_t)(m0 + l15) * HID;
    const bf16_t* bBase = w2t  + (size_t)(n0 + l15) * HID;

#pragma unroll
    for (int k0 = 0; k0 < HID; k0 += 32) {
        // ---- A fragments: lanes 0-15 -> K = k0+{0..7,16..23}; lanes 16-31 +8
        const int ka = k0 + half * 8;
        v16bf a[4];
#pragma unroll
        for (int mt = 0; mt < 4; ++mt) {
            const bf16_t* ap = aBase + (size_t)(mt * 16) * HID + ka;
            const v8bf alo = *(const v8bf*)(ap);
            const v8bf ahi = *(const v8bf*)(ap + 16);
#pragma unroll
            for (int i = 0; i < 8; ++i) { a[mt][i] = alo[i]; a[mt][8 + i] = ahi[i]; }
        }

        // ---- B fragments: lane holds column n, K = k0+{0..7,16..23}; lanes 16-31 +8
        const int kb = k0 + half * 8;
        v16bf b[4];
#pragma unroll
        for (int nt = 0; nt < 4; ++nt) {
            const bf16_t* bp = bBase + (size_t)(nt * 16) * HID + kb;
            const v8bf blo = *(const v8bf*)(bp);
            const v8bf bhi = *(const v8bf*)(bp + 16);
#pragma unroll
            for (int i = 0; i < 8; ++i) { b[nt][i] = blo[i]; b[nt][8 + i] = bhi[i]; }
        }

        // ---- 16 matrix ops per K-step
#pragma unroll
        for (int mt = 0; mt < 4; ++mt) {
#pragma unroll
            for (int nt = 0; nt < 4; ++nt) {
                acc[mt][nt] = __builtin_amdgcn_wmma_f32_16x16x32_bf16(
                    /*neg_a=*/false, a[mt], /*neg_b=*/false, b[nt],
                    /*c_mod=*/(short)0, acc[mt][nt],
                    /*reuse_a=*/false, /*reuse_b=*/false);
            }
        }
    }

    // Epilogue: out = acc*ls + b2*ls  (history is added afterwards)
    // C/D layout: VGPR r -> M = half*8 + r ; N = l15 (within tile)
#pragma unroll
    for (int nt = 0; nt < 4; ++nt) {
        const int n = n0 + nt * 16 + l15;
        if (n < VOC) {
            const float bias = b2[n] * ls;
#pragma unroll
            for (int mt = 0; mt < 4; ++mt) {
#pragma unroll
                for (int r = 0; r < 8; ++r) {
                    const int m = m0 + mt * 16 + half * 8 + r;
                    out[(size_t)m * VOC + n] = acc[mt][nt][r] * ls + bias;
                }
            }
        }
    }
}

// ---------------------------------------------------------------------------
// Kernel 4: deterministic history scatter-add on top of the logits.
// One thread per batch row -> exclusive ownership, no atomics, replay-stable.
// ---------------------------------------------------------------------------
__global__ void __launch_bounds__(256)
history_scatter_kernel(const int* __restrict__ loc_seq,
                       const int* __restrict__ mask,
                       const float* __restrict__ rw_p,
                       const float* __restrict__ fw_p,
                       const float* __restrict__ hs_p,
                       float* __restrict__ out)
{
    const int b = blockIdx.x * blockDim.x + threadIdx.x;
    if (b >= B_) return;

    const float rw = *rw_p;
    const float fw = *fw_p;
    const float hs = *hs_p;

    const int* lrow = loc_seq + (size_t)b * L_;
    const int* mrow = mask    + (size_t)b * L_;
    float* orow = out + (size_t)b * VOC;

    for (int t = 0; t < L_; ++t) {
        if (mrow[t]) {
            const float recency = __expf(-0.1f * (float)(L_ - 1 - t)) * rw;
            const float val = (recency + fw) * hs;
            const int v = lrow[t];
            orow[v] += val;          // serial within this thread: deterministic
        }
    }
}

// ---------------------------------------------------------------------------
extern "C" void kernel_launch(void* const* d_in, const int* in_sizes, int n_in,
                              void* d_out, int out_size, void* d_ws, size_t ws_size,
                              hipStream_t stream) {
    const float* hidden = (const float*)d_in[0];
    const int*   loc    = (const int*)  d_in[1];
    const int*   mask   = (const int*)  d_in[2];
    const float* W1     = (const float*)d_in[3];
    const float* b1     = (const float*)d_in[4];
    const float* gamma  = (const float*)d_in[5];
    const float* beta   = (const float*)d_in[6];
    const float* W2     = (const float*)d_in[7];
    const float* b2     = (const float*)d_in[8];
    const float* rw     = (const float*)d_in[9];
    const float* fw     = (const float*)d_in[10];
    const float* hs     = (const float*)d_in[11];
    const float* ls     = (const float*)d_in[12];
    float* out = (float*)d_out;

    // Workspace layout: [hact bf16 2048*256] [W2t bf16 VPAD*256]
    bf16_t* hact = (bf16_t*)d_ws;
    bf16_t* w2t  = (bf16_t*)((char*)d_ws + (size_t)B_ * HID * sizeof(bf16_t));

    // 1) h = GELU(LN(hidden @ W1 + b1)) -> bf16
    mlp_ln_gelu_kernel<<<B_, 256, 0, stream>>>(hidden, W1, b1, gamma, beta, hact);

    // 2) W2 -> bf16, transposed to (VPAD x 256)
    {
        dim3 grid(VPAD / 32, HID / 32);
        dim3 block(32, 8);
        w2_transpose_bf16_kernel<<<grid, block, 0, stream>>>(W2, w2t);
    }

    // 3) logits = (hact @ W2t^T)*ls + b2*ls  via v_wmma_f32_16x16x32_bf16
    //    64x512 block tile, 64x64 per wave (16 accumulators)
    {
        dim3 grid(VPAD / 512, B_ / 64);
        wmma_logits_kernel<<<grid, 256, 0, stream>>>(hact, w2t, b2, ls, out);
    }

    // 4) out += history scores (deterministic, one thread per row)
    history_scatter_kernel<<<B_ / 256, 256, 0, stream>>>(loc, mask, rw, fw, hs, out);
}
